// GTCN_block_53626961658084
// MI455X (gfx1250) — matrix operand order repaired and checked by reading, not compiled, not run on hardware
//
#include <hip/hip_runtime.h>
#include <hip/hip_bf16.h>

// ---------------------------------------------------------------------------
// ST-GCN block (GCN -> temporal conv -> BN -> ReLU -> residual) for gfx1250.
// All matmuls on v_wmma_f32_16x16x32_bf16 (f32 accumulate).
// TCN stages each (nm,v) temporal sequence into LDS via async global->LDS.
// ---------------------------------------------------------------------------

typedef __attribute__((ext_vector_type(16))) __bf16 v16bf;
typedef __attribute__((ext_vector_type(8)))  float  v8f;

#define WMMA_BF16(a, b, c)                                                     \
  __builtin_amdgcn_wmma_f32_16x16x32_bf16(false, (a), false, (b), (short)0,    \
                                          (c), false, false)

#define USE_ASYNC_LDS 1   // flip to 0 if global_load_async_to_lds_* rejected

// Problem constants
constexpr int VJ   = 25;    // real joints
constexpr int TT   = 300;   // frames
constexpr int CC   = 64;    // channels (in == out)
constexpr int NMB  = 32;    // N*M
constexpr int NFR  = NMB * TT;   // 9600 frames
constexpr int KT   = 9;
constexpr int TTIL = 19;    // ceil(300/16)

// Workspace byte offsets (16B-aligned where vector loads occur)
constexpr size_t OFF_AHAT  = 0;        // 2*32*16 bf16      = 2048 B
constexpr size_t OFF_WSWZ  = 2048;     // 8*32*16 bf16      = 8192 B
constexpr size_t OFF_WC    = 10240;    // 9*8*32*16 bf16    = 73728 B
constexpr size_t OFF_SCALE = 83968;    // 64 f32
constexpr size_t OFF_SHIFT = 84224;    // 64 f32
constexpr size_t OFF_XBUF  = 84480;    // 32*25*300*64 bf16 = 30.72 MB

__device__ __forceinline__ unsigned short f2bf(float f) {
  unsigned u = __builtin_bit_cast(unsigned, f);
  unsigned r = u + 0x7FFFu + ((u >> 16) & 1u);   // round-to-nearest-even
  return (unsigned short)(r >> 16);
}

// Pack two f32 -> bf16x2 in one dword (v_cvt_pk_bf16_f32 when available)
__device__ __forceinline__ unsigned f2bf2(float lo, float hi) {
#if __has_builtin(__builtin_amdgcn_cvt_pk_bf16_f32)
  typedef __attribute__((ext_vector_type(2))) __bf16 v2bf;
  v2bf r = __builtin_amdgcn_cvt_pk_bf16_f32(lo, hi);
  return __builtin_bit_cast(unsigned, r);
#else
  return (unsigned)f2bf(lo) | ((unsigned)f2bf(hi) << 16);
#endif
}

union FragB {
  uint4 q[2];
  v16bf v;
  unsigned d[8];
  unsigned short s[16];
};

// ---------------------------------------------------------------------------
// Prep: build A_hat, swizzle weights into WMMA fragment layout, fold BN.
// Fragment layout (per lane, 16 contiguous bf16):
//   A_hat 16x32:  K = hi*16 + j  (row = ln)
//   B 32x16:  j<8 -> K = hi*8+j ; j>=8 -> K = 16+hi*8+(j-8)                                   (col = ln)
// ---------------------------------------------------------------------------
__global__ void __launch_bounds__(256) prep_kernel(
    const float* __restrict__ adj, const float* __restrict__ gcn_w,
    const float* __restrict__ conv_w, const float* __restrict__ conv_b,
    const float* __restrict__ gamma, const float* __restrict__ beta,
    const float* __restrict__ mean, const float* __restrict__ var,
    unsigned short* __restrict__ ahat, unsigned short* __restrict__ wswz,
    unsigned short* __restrict__ wc, float* __restrict__ scale,
    float* __restrict__ shift) {
  __shared__ float s_norm[VJ];
  const int tid = threadIdx.x;
  if (tid < VJ) {
    float d = 0.f;
    for (int u = 0; u < VJ; ++u) d += adj[tid * VJ + u];
    s_norm[tid] = rsqrtf(d);
  }
  __syncthreads();

  // A_hat fragments: idx = ((mi*32 + lane)*16 + j)
  for (int idx = tid; idx < 2 * 32 * 16; idx += 256) {
    const int j = idx & 15, lane = (idx >> 4) & 31, mi = idx >> 9;
    const int ln = lane & 15, hi = lane >> 4;
    const int m = mi * 16 + ln;
    const int K = hi * 16 + j;
    float val = 0.f;
    if (m < VJ && K < VJ) val = s_norm[m] * adj[m * VJ + K] * s_norm[K];
    ahat[idx] = f2bf(val);
  }
  // gcn_w B fragments: idx = (((ks*4+nt)*32 + lane)*16 + j)
  for (int idx = tid; idx < 8 * 32 * 16; idx += 256) {
    const int j = idx & 15, lane = (idx >> 4) & 31, f = idx >> 9;
    const int ks = f >> 2, nt = f & 3;
    const int ln = lane & 15, hi = lane >> 4;
    const int K = ks * 32 + ((j < 8) ? (hi * 8 + j) : (16 + hi * 8 + (j - 8)));       // ci
    const int n = nt * 16 + ln;                // co
    wswz[idx] = f2bf(gcn_w[K * CC + n]);
  }
  // conv_w B fragments: idx = ((((k*8)+(ks*4+nt))*32 + lane)*16 + j)
  for (int idx = tid; idx < KT * 8 * 32 * 16; idx += 256) {
    const int j = idx & 15, lane = (idx >> 4) & 31;
    const int f = (idx >> 9) & 7, k = idx >> 12;
    const int ks = f >> 2, nt = f & 3;
    const int ln = lane & 15, hi = lane >> 4;
    const int ci = ks * 32 + ((j < 8) ? (hi * 8 + j) : (16 + hi * 8 + (j - 8)));
    const int co = nt * 16 + ln;
    wc[idx] = f2bf(conv_w[(co * CC + ci) * KT + k]);  // (O,I,KT,1)
  }
  if (tid < CC) {
    const float s = gamma[tid] * rsqrtf(var[tid] + 1e-5f);
    scale[tid] = s;
    shift[tid] = (conv_b[tid] - mean[tid]) * s + beta[tid];
  }
}

// ---------------------------------------------------------------------------
// GCN: per frame b=(nm,t):  X1 = h_b @ W (32x64, rows>=25 zero)
//      X2 = relu(A_hat @ X1 + bias)   -> xbuf[nm][v][t][c] (bf16)
// One wave per frame; D-frags of GEMM1 bounce through LDS transposed [c][u]
// so GEMM2's B-frags are contiguous 16B ds loads.
// ---------------------------------------------------------------------------
__global__ void __launch_bounds__(256) gcn_kernel(
    const float* __restrict__ h, const unsigned short* __restrict__ ahat,
    const unsigned short* __restrict__ wswz, const float* __restrict__ gcn_b,
    unsigned short* __restrict__ xbuf) {
  __shared__ __align__(16) unsigned short x1t[8][CC * 32];  // per-wave 4KB
  const int wv = threadIdx.x >> 5;
  const int lane = threadIdx.x & 31;
  const int ln = lane & 15, hi = lane >> 4;
  const int frame = blockIdx.x * 8 + wv;          // 0..9599 (exact)
  const int nm = frame / TT, t = frame - nm * TT;
  const float* hbase = h + (size_t)frame * VJ * CC;

  v8f acc[2][4] = {};

  // GEMM1: X1 = h @ W   (M=u joints, K=ci, N=c)
#pragma unroll
  for (int ks = 0; ks < 2; ++ks) {
    v16bf a[2];
#pragma unroll
    for (int mi = 0; mi < 2; ++mi) {
      const int u = mi * 16 + ln;
      FragB fa;
      if (u < VJ) {
        const float* row = hbase + u * CC + ks * 32 + hi * 8;
        const float4 f0 = *(const float4*)(row + 0);
        const float4 f1 = *(const float4*)(row + 4);
        const float4 f2 = *(const float4*)(row + 16);
        const float4 f3 = *(const float4*)(row + 20);
        fa.d[0] = f2bf2(f0.x, f0.y);
        fa.d[1] = f2bf2(f0.z, f0.w);
        fa.d[2] = f2bf2(f1.x, f1.y);
        fa.d[3] = f2bf2(f1.z, f1.w);
        fa.d[4] = f2bf2(f2.x, f2.y);
        fa.d[5] = f2bf2(f2.z, f2.w);
        fa.d[6] = f2bf2(f3.x, f3.y);
        fa.d[7] = f2bf2(f3.z, f3.w);
      } else {
        fa.q[0] = make_uint4(0, 0, 0, 0);
        fa.q[1] = make_uint4(0, 0, 0, 0);
      }
      a[mi] = fa.v;
    }
#pragma unroll
    for (int nt = 0; nt < 4; ++nt) {
      const unsigned short* wp = wswz + ((ks * 4 + nt) * 32 + lane) * 16;
      FragB fb;
      fb.q[0] = *(const uint4*)wp;
      fb.q[1] = *(const uint4*)(wp + 8);
      acc[0][nt] = WMMA_BF16(a[0], fb.v, acc[0][nt]);
      acc[1][nt] = WMMA_BF16(a[1], fb.v, acc[1][nt]);
    }
  }

  // Store X1^T into LDS: [c][u] so each lane writes 8 contiguous bf16 (16B)
  unsigned short* lw = x1t[wv];
#pragma unroll
  for (int mi = 0; mi < 2; ++mi)
#pragma unroll
    for (int nt = 0; nt < 4; ++nt) {
      const int c = nt * 16 + ln;
      const int u0 = mi * 16 + hi * 8;
      uint4 tv;
      tv.x = f2bf2(acc[mi][nt][0], acc[mi][nt][1]);
      tv.y = f2bf2(acc[mi][nt][2], acc[mi][nt][3]);
      tv.z = f2bf2(acc[mi][nt][4], acc[mi][nt][5]);
      tv.w = f2bf2(acc[mi][nt][6], acc[mi][nt][7]);
      *(uint4*)(lw + c * 32 + u0) = tv;
    }
  asm volatile("s_wait_dscnt 0" ::: "memory");  // same-wave LDS RAW

  // GEMM2: X2 = A_hat @ X1  (K=u=32, one WMMA per tile), bias in accumulator
  v16bf ah[2];
#pragma unroll
  for (int mi = 0; mi < 2; ++mi) {
    const unsigned short* ap = ahat + (mi * 32 + lane) * 16;
    FragB fa;
    fa.q[0] = *(const uint4*)ap;
    fa.q[1] = *(const uint4*)(ap + 8);
    ah[mi] = fa.v;
  }
#pragma unroll
  for (int nt = 0; nt < 4; ++nt) {
    const int c = nt * 16 + ln;
    const unsigned short* xp = lw + c * 32 + hi * 16;
    FragB fb;
    fb.q[0] = *(const uint4*)xp;
    fb.q[1] = *(const uint4*)(xp + 8);
    const float bv = gcn_b[c];
    const v8f cinit = {bv, bv, bv, bv, bv, bv, bv, bv};
#pragma unroll
    for (int mi = 0; mi < 2; ++mi) {
      const v8f d = WMMA_BF16(ah[mi], fb.v, cinit);
#pragma unroll
      for (int r = 0; r < 8; ++r) {
        const int v = mi * 16 + hi * 8 + r;
        if (v < VJ) {
          float val = d[r];
          val = val > 0.f ? val : 0.f;
          xbuf[((size_t)(nm * VJ + v) * TT + t) * CC + c] = f2bf(val);
        }
      }
    }
  }
}

// ---------------------------------------------------------------------------
// TCN: out[t,co] = relu( (sum_k x[t+k-4,:] @ Wk)*scale + shift ) + h
// One WG per (nm,v): stage the 300x64 bf16 sequence (37.5KB) into LDS once
// via async global->LDS, then 8 waves cover t-tiles wv, wv+8, wv+16.
// 72 WMMA per tile; A-frags are contiguous 16B LDS reads.
// ---------------------------------------------------------------------------
__global__ void __launch_bounds__(256) tcn_kernel(
    const float* __restrict__ h, const unsigned short* __restrict__ xbuf,
    const unsigned short* __restrict__ wc, const float* __restrict__ scale,
    const float* __restrict__ shift, float* __restrict__ out) {
  extern __shared__ __align__(16) unsigned short xs[];   // TT*CC bf16
  const int wv = threadIdx.x >> 5;
  const int lane = threadIdx.x & 31;
  const int ln = lane & 15, hi = lane >> 4;
  const int nmv = blockIdx.x;                 // 0..799
  const int v = nmv % VJ, nm = nmv / VJ;
  const unsigned short* xrow = xbuf + (size_t)nmv * TT * CC;

  // ---- stage sequence into LDS (2400 x 16B chunks) ----
  {
    const unsigned lds0 = (unsigned)(size_t)xs;   // LDS byte address (low 32b)
#if USE_ASYNC_LDS
    for (int i = threadIdx.x; i < (TT * CC * 2) / 16; i += 256) {
      const void* g = (const char*)xrow + i * 16;
      asm volatile("global_load_async_to_lds_b128 %0, %1, off"
                   :: "v"(lds0 + i * 16), "v"(g)
                   : "memory");
    }
    asm volatile("s_wait_asynccnt 0" ::: "memory");
#else
    for (int i = threadIdx.x; i < (TT * CC * 2) / 16; i += 256) {
      *(uint4*)((char*)xs + i * 16) = *(const uint4*)((const char*)xrow + i * 16);
    }
#endif
  }
  __syncthreads();

  for (int tt = wv; tt < TTIL; tt += 8) {
    const int t0 = tt * 16;
    v8f acc[4] = {};
#pragma unroll
    for (int k = 0; k < KT; ++k) {
      const int tr = t0 + k - 4 + ln;               // input frame, this lane
      const bool ok = (unsigned)tr < (unsigned)TT;  // zero-pad halo
      v16bf a[2];
#pragma unroll
      for (int ks = 0; ks < 2; ++ks) {
        FragB fa;
        if (ok) {
          const unsigned short* p = xs + tr * CC + ks * 32 + hi * 8;
          fa.q[0] = *(const uint4*)p;          // K = base + 0..7
          fa.q[1] = *(const uint4*)(p + 16);   // K = base + 16..23
        } else {
          fa.q[0] = make_uint4(0, 0, 0, 0);
          fa.q[1] = make_uint4(0, 0, 0, 0);
        }
        a[ks] = fa.v;
      }
      const unsigned short* wk = wc + (size_t)k * 8 * 32 * 16;
#pragma unroll
      for (int nt = 0; nt < 4; ++nt)
#pragma unroll
        for (int ks = 0; ks < 2; ++ks) {
          const unsigned short* wp = wk + ((ks * 4 + nt) * 32 + lane) * 16;
          FragB fb;
          fb.q[0] = *(const uint4*)wp;
          fb.q[1] = *(const uint4*)(wp + 8);
          acc[nt] = WMMA_BF16(a[ks], fb.v, acc[nt]);
        }
    }

    // Epilogue: BN (folded) + ReLU + residual, f32 out
#pragma unroll
    for (int nt = 0; nt < 4; ++nt) {
      const int c = nt * 16 + ln;
      const float sc = scale[c], sh = shift[c];
#pragma unroll
      for (int r = 0; r < 8; ++r) {
        const int t = t0 + hi * 8 + r;
        if (t < TT) {
          const size_t idx = ((size_t)(nm * TT + t) * VJ + v) * CC + c;
          float val = acc[nt][r] * sc + sh;
          val = val > 0.f ? val : 0.f;
          out[idx] = val + h[idx];
        }
      }
    }
  }
}

// ---------------------------------------------------------------------------
extern "C" void kernel_launch(void* const* d_in, const int* in_sizes, int n_in,
                              void* d_out, int out_size, void* d_ws,
                              size_t ws_size, hipStream_t stream) {
  const float* h      = (const float*)d_in[0];
  const float* adj    = (const float*)d_in[1];
  const float* gcn_w  = (const float*)d_in[2];
  const float* gcn_b  = (const float*)d_in[3];
  const float* conv_w = (const float*)d_in[4];
  const float* conv_b = (const float*)d_in[5];
  const float* gamma  = (const float*)d_in[6];
  const float* beta   = (const float*)d_in[7];
  const float* mean   = (const float*)d_in[8];
  const float* var    = (const float*)d_in[9];

  char* ws = (char*)d_ws;
  unsigned short* ahat  = (unsigned short*)(ws + OFF_AHAT);
  unsigned short* wswz  = (unsigned short*)(ws + OFF_WSWZ);
  unsigned short* wc    = (unsigned short*)(ws + OFF_WC);
  float*          scale = (float*)(ws + OFF_SCALE);
  float*          shift = (float*)(ws + OFF_SHIFT);
  unsigned short* xbuf  = (unsigned short*)(ws + OFF_XBUF);

  prep_kernel<<<1, 256, 0, stream>>>(adj, gcn_w, conv_w, conv_b, gamma, beta,
                                     mean, var, ahat, wswz, wc, scale, shift);
  gcn_kernel<<<NFR / 8, 256, 0, stream>>>(h, ahat, wswz, gcn_b, xbuf);
  tcn_kernel<<<NMB * VJ, 256, TT * CC * 2, stream>>>(h, xbuf, wc, scale, shift,
                                                     (float*)d_out);
}
